// NonLocalASPP_42314017800767
// MI455X (gfx1250) — matrix operand order repaired and checked by reading, not compiled, not run on hardware
//
#include <hip/hip_runtime.h>
#include <hip/hip_bf16.h>

typedef __bf16 bf16;
typedef __attribute__((ext_vector_type(16))) __bf16 v16bf;
typedef __attribute__((ext_vector_type(8)))  __bf16 v8bf;
typedef __attribute__((ext_vector_type(8)))  float  v8f;

#define BB 4
#define CC 64
#define CIH 32
#define NN 4096
#define DD 64
#define HH 64
#define WW 64

// ---------------------------------------------------------------- WMMA core
__device__ __forceinline__ v8f wmma_bf(v16bf a, v16bf b, v8f c) {
    // v_wmma_f32_16x16x32_bf16: D = A(16x32) * B(32x16) + C(16x16 f32)
    return __builtin_amdgcn_wmma_f32_16x16x32_bf16(false, a, false, b,
                                                   (short)0, c, false, false);
}

__device__ __forceinline__ v8f zero8() {
    v8f z;
#pragma unroll
    for (int i = 0; i < 8; ++i) z[i] = 0.0f;
    return z;
}

__device__ __forceinline__ v16bf mk16(v8bf lo, v8bf hi) {
    v16bf r;
#pragma unroll
    for (int i = 0; i < 8; ++i) { r[i] = lo[i]; r[i + 8] = hi[i]; }
    return r;
}

// A fragment 16x32 bf16. Source: row-major [16 rows][K], p -> (row0, K0).
// ISA layout: lane<16 holds row=lane, K = {8h..8h+7} U {16+8h..23+8h}, h=lane>>4.
__device__ __forceinline__ v16bf load_a(const bf16* p, int ldk, int lane) {
    int half = lane >> 4;
    const bf16* r = p + (lane & 15) * ldk + (half << 3);
    return mk16(*(const v8bf*)r, *(const v8bf*)(r + 16));
}

// B fragment 32x16 bf16 (KxN). Source: [N rows][K] row-major (weight [out][in]).
// ISA layout: lane holds col N=lane&15, K = {8h..8h+7} U {16+8h..23+8h}, h=lane>>4.
__device__ __forceinline__ v16bf load_b(const bf16* p, int ldk, int lane) {
    int half = lane >> 4;
    const bf16* r = p + (lane & 15) * ldk + (half << 3);
    return mk16(*(const v8bf*)r, *(const v8bf*)(r + 16));
}

// A fragment from zt with spatial shift (dilated conv tap), zero-padded OOB.
__device__ __forceinline__ v16bf load_a_dil(const bf16* zb, int h, int w0,
                                            int dy, int dx, int K0, int lane) {
    int half = lane >> 4;
    int w  = w0 + (lane & 15) + dx;
    int h2 = h + dy;
    v8bf lo, hi;
#pragma unroll
    for (int i = 0; i < 8; ++i) { lo[i] = (bf16)0.0f; hi[i] = (bf16)0.0f; }
    if ((unsigned)h2 < (unsigned)HH && (unsigned)w < (unsigned)WW) {
        const bf16* r = zb + ((size_t)(h2 * WW + w)) * CC + K0 + (half << 3);
        lo = *(const v8bf*)r;
        hi = *(const v8bf*)(r + 16);
    }
    return mk16(lo, hi);
}

__device__ __forceinline__ float redmax16(float v) {
#pragma unroll
    for (int s = 1; s < 16; s <<= 1) v = fmaxf(v, __shfl_xor(v, s, 32));
    return v;
}
__device__ __forceinline__ float redsum16(float v) {
#pragma unroll
    for (int s = 1; s < 16; s <<= 1) v += __shfl_xor(v, s, 32);
    return v;
}

// ---------------------------------------------------------------- prep
__global__ void cvt_kernel(const float* __restrict__ src, bf16* __restrict__ dst, int count) {
    int i = blockIdx.x * 256 + threadIdx.x;
    if (i < count) dst[i] = (bf16)src[i];
}

__global__ void transpose_x_kernel(const float* __restrict__ x, bf16* __restrict__ xt) {
    int i = blockIdx.x * 256 + threadIdx.x;           // over B*C*N
    if (i < BB * CC * NN) {
        int n = i & (NN - 1);
        int c = (i >> 12) & (CC - 1);
        int b = i >> 18;
        xt[((size_t)b * NN + n) * CC + c] = (bf16)x[i];
    }
}

// a{3,5,7}_w [d][c][3][3] f32 -> [tap][d][c] bf16
__global__ void tapcvt_kernel(const float* __restrict__ src, bf16* __restrict__ dst) {
    int i = blockIdx.x * 256 + threadIdx.x;           // over D*C*9
    if (i < DD * CC * 9) {
        int t = i % 9;
        int c = (i / 9) % CC;
        int d = i / (9 * CC);
        dst[(size_t)t * DD * CC + d * CC + c] = (bf16)src[i];
    }
}

// ---------------------------------------------------------------- projections
__global__ void __launch_bounds__(128) proj_kernel(
    const bf16* __restrict__ xt,
    const bf16* __restrict__ wth, const bf16* __restrict__ wph, const bf16* __restrict__ wg,
    const float* __restrict__ bth, const float* __restrict__ bph, const float* __restrict__ bg,
    bf16* __restrict__ th, bf16* __restrict__ ph, bf16* __restrict__ gt)
{
    int lane = threadIdx.x & 31;
    int task = blockIdx.x * 4 + (threadIdx.x >> 5);   // B*256 tasks
    int b = task >> 8, n0 = (task & 255) << 4;
    const bf16* xr = xt + ((size_t)b * NN + n0) * CC;
    v16bf a0 = load_a(xr, CC, lane);
    v16bf a1 = load_a(xr + 32, CC, lane);
    int nn = lane & 15, half = lane >> 4;
#pragma unroll
    for (int t = 0; t < 2; ++t) {
        int ci = nn + t * 16;
        // theta
        v8f acc = wmma_bf(a0, load_b(wth + (t * 16) * CC, CC, lane), zero8());
        acc = wmma_bf(a1, load_b(wth + (t * 16) * CC + 32, CC, lane), acc);
        float bias = bth[ci];
#pragma unroll
        for (int i = 0; i < 8; ++i) {
            int n = n0 + i + 8 * half;
            th[((size_t)b * NN + n) * CIH + ci] = (bf16)(acc[i] + bias);
        }
        // phi
        acc = wmma_bf(a0, load_b(wph + (t * 16) * CC, CC, lane), zero8());
        acc = wmma_bf(a1, load_b(wph + (t * 16) * CC + 32, CC, lane), acc);
        bias = bph[ci];
#pragma unroll
        for (int i = 0; i < 8; ++i) {
            int n = n0 + i + 8 * half;
            ph[((size_t)b * NN + n) * CIH + ci] = (bf16)(acc[i] + bias);
        }
        // g, stored transposed [ci][n] so it is B-fragment-ready for O = P*G
        acc = wmma_bf(a0, load_b(wg + (t * 16) * CC, CC, lane), zero8());
        acc = wmma_bf(a1, load_b(wg + (t * 16) * CC + 32, CC, lane), acc);
        bias = bg[ci];
#pragma unroll
        for (int i = 0; i < 8; ++i) {
            int n = n0 + i + 8 * half;
            gt[(size_t)b * CIH * NN + (size_t)ci * NN + n] = (bf16)(acc[i] + bias);
        }
    }
}

// ------------------------------------------------ flash attention + Wz + BN + residual
__global__ void __launch_bounds__(128) attn_kernel(
    const bf16* __restrict__ th, const bf16* __restrict__ ph, const bf16* __restrict__ gt,
    const bf16* __restrict__ wz, const float* __restrict__ wz_b,
    const float* __restrict__ bn_g, const float* __restrict__ bn_b,
    const bf16* __restrict__ xt, bf16* __restrict__ zt)
{
    __shared__ __attribute__((aligned(16))) bf16 pshare[4][16 * 32];
    int lane = threadIdx.x & 31, wv = threadIdx.x >> 5;
    int task = blockIdx.x * 4 + wv;                   // B*256 query blocks
    int b = task >> 8, q0 = (task & 255) << 4;
    bf16* P = &pshare[wv][0];
    const bf16* phb = ph + (size_t)b * NN * CIH;
    const bf16* gtb = gt + (size_t)b * CIH * NN;
    int nn = lane & 15, half = lane >> 4;

    v16bf qa = load_a(th + ((size_t)b * NN + q0) * CIH, CIH, lane);

    float m[8], l[8];
    v8f o0 = zero8(), o1 = zero8();
#pragma unroll
    for (int i = 0; i < 8; ++i) { m[i] = -1.0e30f; l[i] = 0.0f; }

    for (int k0 = 0; k0 < NN; k0 += 32) {
        v8f s0 = wmma_bf(qa, load_b(phb + (size_t)k0 * CIH, CIH, lane), zero8());
        v8f s1 = wmma_bf(qa, load_b(phb + (size_t)(k0 + 16) * CIH, CIH, lane), zero8());
#pragma unroll
        for (int i = 0; i < 8; ++i) {
            float bm = redmax16(fmaxf(s0[i], s1[i]));
            float mn = fmaxf(m[i], bm);
            float sc = __expf(m[i] - mn);
            float p0 = __expf(s0[i] - mn);
            float p1 = __expf(s1[i] - mn);
            l[i] = l[i] * sc + redsum16(p0 + p1);
            o0[i] *= sc;
            o1[i] *= sc;
            m[i] = mn;
            int M = i + 8 * half;                     // C-frag row
            P[M * 32 + nn]      = (bf16)p0;
            P[M * 32 + 16 + nn] = (bf16)p1;
        }
        v16bf pa = load_a(P, 32, lane);               // P re-enters as A fragment
        o0 = wmma_bf(pa, load_b(gtb + k0, NN, lane), o0);
        o1 = wmma_bf(pa, load_b(gtb + (size_t)16 * NN + k0, NN, lane), o1);
    }

    // normalize and stage Y (16 x 32ci) in LDS for the Wz GEMM
#pragma unroll
    for (int i = 0; i < 8; ++i) {
        float inv = 1.0f / l[i];
        int M = i + 8 * half;
        P[M * 32 + nn]      = (bf16)(o0[i] * inv);
        P[M * 32 + 16 + nn] = (bf16)(o1[i] * inv);
    }
    v16bf ya = load_a(P, 32, lane);
    const float inv_std = rsqrtf(1.0f + 1e-5f);
#pragma unroll
    for (int t = 0; t < 4; ++t) {
        v8f zf = wmma_bf(ya, load_b(wz + (t * 16) * CIH, CIH, lane), zero8());
        int c = nn + t * 16;
        float s  = bn_g[c] * inv_std;
        float wb = wz_b[c];
        float bb = bn_b[c];
#pragma unroll
        for (int i = 0; i < 8; ++i) {
            int n = q0 + i + 8 * half;
            float xv = (float)xt[((size_t)b * NN + n) * CC + c];
            zt[((size_t)b * NN + n) * CC + c] = (bf16)((zf[i] + wb) * s + bb + xv);
        }
    }
}

// ---------------------------------------------------------------- mean + img branch
__global__ void meanimg_kernel(const bf16* __restrict__ zt,
                               const float* __restrict__ amw, const float* __restrict__ amb,
                               bf16* __restrict__ cat)
{
    __shared__ float red[256];
    __shared__ bf16 img[64];
    int b = blockIdx.x, tid = threadIdx.x;
    int c = tid & 63, seg = tid >> 6;
    const bf16* zb = zt + (size_t)b * NN * CC;
    float s = 0.0f;
    for (int p = seg * 1024; p < (seg + 1) * 1024; ++p)
        s += (float)zb[(size_t)p * CC + c];
    red[tid] = s;
    __syncthreads();
    if (tid < 64)
        red[c] = (red[c] + red[c + 64] + red[c + 128] + red[c + 192]) * (1.0f / (float)NN);
    __syncthreads();
    if (tid < 64) {
        float v = amb[tid];
        for (int c2 = 0; c2 < CC; ++c2) v += amw[tid * CC + c2] * red[c2];
        img[tid] = (bf16)v;
    }
    __syncthreads();
    for (int n = tid; n < NN; n += 256) {
        bf16* dst = cat + ((size_t)b * NN + n) * 320;
#pragma unroll
        for (int d = 0; d < 64; ++d) dst[d] = img[d];
    }
}

// ---------------------------------------------------------------- ASPP conv branches
__global__ void __launch_bounds__(128) branch_kernel(
    const bf16* __restrict__ zt,
    const bf16* __restrict__ wa1, const bf16* __restrict__ wt3,
    const bf16* __restrict__ wt5, const bf16* __restrict__ wt7,
    const float* __restrict__ b1b, const float* __restrict__ b3b,
    const float* __restrict__ b5b, const float* __restrict__ b7b,
    bf16* __restrict__ cat)
{
    int lane = threadIdx.x & 31;
    int task = blockIdx.x * 4 + (threadIdx.x >> 5);   // B*4*256 tasks
    int pb = task & 255, br = (task >> 8) & 3, b = task >> 10;
    int n0 = pb << 4, h = n0 >> 6, w0 = n0 & 63;
    const bf16* zb = zt + (size_t)b * NN * CC;
    v8f acc[4];
#pragma unroll
    for (int t = 0; t < 4; ++t) acc[t] = zero8();
    const float* bias;
    if (br == 0) {                                    // 1x1 conv
        v16bf a0 = load_a(zb + (size_t)n0 * CC, CC, lane);
        v16bf a1 = load_a(zb + (size_t)n0 * CC + 32, CC, lane);
#pragma unroll
        for (int t = 0; t < 4; ++t) {
            acc[t] = wmma_bf(a0, load_b(wa1 + (t * 16) * CC, CC, lane), acc[t]);
            acc[t] = wmma_bf(a1, load_b(wa1 + (t * 16) * CC + 32, CC, lane), acc[t]);
        }
        bias = b1b;
    } else {                                          // dilated 3x3 as 9 tap-GEMMs
        int dil = 2 * br + 1;
        const bf16* tw = (br == 1) ? wt3 : ((br == 2) ? wt5 : wt7);
        bias            = (br == 1) ? b3b : ((br == 2) ? b5b : b7b);
        for (int tap = 0; tap < 9; ++tap) {
            int dy = (tap / 3 - 1) * dil, dx = (tap % 3 - 1) * dil;
            v16bf a0 = load_a_dil(zb, h, w0, dy, dx, 0, lane);
            v16bf a1 = load_a_dil(zb, h, w0, dy, dx, 32, lane);
            const bf16* twt = tw + (size_t)tap * DD * CC;
#pragma unroll
            for (int t = 0; t < 4; ++t) {
                acc[t] = wmma_bf(a0, load_b(twt + (t * 16) * CC, CC, lane), acc[t]);
                acc[t] = wmma_bf(a1, load_b(twt + (t * 16) * CC + 32, CC, lane), acc[t]);
            }
        }
    }
    int col0 = 64 + 64 * br;
    int nn = lane & 15, half = lane >> 4;
#pragma unroll
    for (int t = 0; t < 4; ++t) {
        int c = nn + t * 16;
        float bv = bias[c];
#pragma unroll
        for (int i = 0; i < 8; ++i) {
            int n = n0 + i + 8 * half;
            cat[((size_t)b * NN + n) * 320 + col0 + c] = (bf16)(acc[t][i] + bv);
        }
    }
}

// ---------------------------------------------------------------- final 320->64 conv
__global__ void __launch_bounds__(128) out_kernel(
    const bf16* __restrict__ cat, const bf16* __restrict__ wout,
    const float* __restrict__ outb, float* __restrict__ out)
{
    int lane = threadIdx.x & 31;
    int task = blockIdx.x * 4 + (threadIdx.x >> 5);   // B*256 tasks
    int b = task >> 8, n0 = (task & 255) << 4;
    const bf16* cr = cat + ((size_t)b * NN + n0) * 320;
    v8f acc[4];
#pragma unroll
    for (int t = 0; t < 4; ++t) acc[t] = zero8();
#pragma unroll
    for (int ks = 0; ks < 10; ++ks) {                 // K = 320
        v16bf a = load_a(cr + ks * 32, 320, lane);
#pragma unroll
        for (int t = 0; t < 4; ++t)
            acc[t] = wmma_bf(a, load_b(wout + (t * 16) * 320 + ks * 32, 320, lane), acc[t]);
    }
    int nn = lane & 15, half = lane >> 4;
#pragma unroll
    for (int t = 0; t < 4; ++t) {
        int d = nn + t * 16;
        float bv = outb[d];
#pragma unroll
        for (int i = 0; i < 8; ++i) {
            int n = n0 + i + 8 * half;
            out[((size_t)(b * DD) + d) * NN + n] = acc[t][i] + bv;
        }
    }
}

// ---------------------------------------------------------------- launcher
extern "C" void kernel_launch(void* const* d_in, const int* in_sizes, int n_in,
                              void* d_out, int out_size, void* d_ws, size_t ws_size,
                              hipStream_t stream) {
    (void)in_sizes; (void)n_in; (void)out_size; (void)ws_size;
    const float* x       = (const float*)d_in[0];
    const float* g_w     = (const float*)d_in[1];
    const float* g_b     = (const float*)d_in[2];
    const float* theta_w = (const float*)d_in[3];
    const float* theta_b = (const float*)d_in[4];
    const float* phi_w   = (const float*)d_in[5];
    const float* phi_b   = (const float*)d_in[6];
    const float* Wz_w    = (const float*)d_in[7];
    const float* Wz_b    = (const float*)d_in[8];
    const float* bn_g    = (const float*)d_in[9];
    const float* bn_b    = (const float*)d_in[10];
    const float* am_w    = (const float*)d_in[11];
    const float* am_b    = (const float*)d_in[12];
    const float* a1_w    = (const float*)d_in[13];
    const float* a1_b    = (const float*)d_in[14];
    const float* a3_w    = (const float*)d_in[15];
    const float* a3_b    = (const float*)d_in[16];
    const float* a5_w    = (const float*)d_in[17];
    const float* a5_b    = (const float*)d_in[18];
    const float* a7_w    = (const float*)d_in[19];
    const float* a7_b    = (const float*)d_in[20];
    const float* out_w   = (const float*)d_in[21];
    const float* out_b   = (const float*)d_in[22];
    float* out = (float*)d_out;

    // workspace carve-up (~18 MB total)
    char* ws = (char*)d_ws;
    size_t off = 0;
    auto alc = [&](size_t bytes) -> size_t {
        size_t r = off;
        off += (bytes + 255) & ~(size_t)255;
        return r;
    };
    bf16* xt   = (bf16*)(ws + alc((size_t)BB * NN * CC * 2));
    bf16* th   = (bf16*)(ws + alc((size_t)BB * NN * CIH * 2));
    bf16* ph   = (bf16*)(ws + alc((size_t)BB * NN * CIH * 2));
    bf16* gt   = (bf16*)(ws + alc((size_t)BB * CIH * NN * 2));
    bf16* zt   = (bf16*)(ws + alc((size_t)BB * NN * CC * 2));
    bf16* cat  = (bf16*)(ws + alc((size_t)BB * NN * 320 * 2));
    bf16* wth  = (bf16*)(ws + alc(CIH * CC * 2));
    bf16* wph  = (bf16*)(ws + alc(CIH * CC * 2));
    bf16* wg   = (bf16*)(ws + alc(CIH * CC * 2));
    bf16* wwz  = (bf16*)(ws + alc(CC * CIH * 2));
    bf16* wa1  = (bf16*)(ws + alc(DD * CC * 2));
    bf16* wt3  = (bf16*)(ws + alc(9 * DD * CC * 2));
    bf16* wt5  = (bf16*)(ws + alc(9 * DD * CC * 2));
    bf16* wt7  = (bf16*)(ws + alc(9 * DD * CC * 2));
    bf16* wout = (bf16*)(ws + alc(DD * 320 * 2));

    // --- prep: bf16 staging in GEMM-friendly layouts
    cvt_kernel<<<(CIH * CC + 255) / 256, 256, 0, stream>>>(theta_w, wth, CIH * CC);
    cvt_kernel<<<(CIH * CC + 255) / 256, 256, 0, stream>>>(phi_w,   wph, CIH * CC);
    cvt_kernel<<<(CIH * CC + 255) / 256, 256, 0, stream>>>(g_w,     wg,  CIH * CC);
    cvt_kernel<<<(CC * CIH + 255) / 256, 256, 0, stream>>>(Wz_w,    wwz, CC * CIH);
    cvt_kernel<<<(DD * CC + 255) / 256, 256, 0, stream>>>(a1_w,     wa1, DD * CC);
    cvt_kernel<<<(DD * 320 + 255) / 256, 256, 0, stream>>>(out_w,   wout, DD * 320);
    tapcvt_kernel<<<(DD * CC * 9 + 255) / 256, 256, 0, stream>>>(a3_w, wt3);
    tapcvt_kernel<<<(DD * CC * 9 + 255) / 256, 256, 0, stream>>>(a5_w, wt5);
    tapcvt_kernel<<<(DD * CC * 9 + 255) / 256, 256, 0, stream>>>(a7_w, wt7);
    transpose_x_kernel<<<(BB * CC * NN) / 256, 256, 0, stream>>>(x, xt);

    // --- nonlocal block
    proj_kernel<<<BB * 256 / 4, 128, 0, stream>>>(xt, wth, wph, wg,
                                                  theta_b, phi_b, g_b, th, ph, gt);
    attn_kernel<<<BB * 256 / 4, 128, 0, stream>>>(th, ph, gt, wwz, Wz_b,
                                                  bn_g, bn_b, xt, zt);

    // --- ASPP
    meanimg_kernel<<<BB, 256, 0, stream>>>(zt, am_w, am_b, cat);
    branch_kernel<<<BB * 4 * 256 / 4, 128, 0, stream>>>(zt, wa1, wt3, wt5, wt7,
                                                        a1_b, a3_b, a5_b, a7_b, cat);
    out_kernel<<<BB * 256 / 4, 128, 0, stream>>>(cat, wout, out_b, out);
}
